// UNetModel_12240656793721
// MI455X (gfx1250) — compile-verified, matrix operand order checked by reading, not run on hardware
//
#include <hip/hip_runtime.h>
#include <hip/hip_bf16.h>
#include <math.h>
#include <stddef.h>

typedef _Float16 h16;
typedef __attribute__((ext_vector_type(16))) _Float16 v16h;
typedef __attribute__((ext_vector_type(8)))  _Float16 v8h;
typedef __attribute__((ext_vector_type(8)))  float    v8f;
typedef __attribute__((ext_vector_type(4)))  unsigned int u32x4;
typedef __attribute__((ext_vector_type(8)))  int      i32x8;
typedef __attribute__((ext_vector_type(4)))  int      i32x4;

#define NGROUPS 32

// -------------------------------------------------------------------------
// All activations: channels-last (NDHWC) f16, index = ((b*VOX + v)*C + c).
// CDNA5 16-bit A/B fragment layout (wave32):
//   lane L<16 : row = L,    halves 0..7 -> K 0..7,  halves 8..15 -> K 16..23
//   lane L>=16: row = L-16, halves 0..7 -> K 8..15, halves 8..15 -> K 24..31
// GEMM K ordering: k = tap*Cin + ci (tap-major, Cin % 32 == 0).
// B fragments are staged into LDS per-block via the Tensor Data Mover
// (tensor_load_to_lds + s_wait_tensorcnt), shared by all 8 waves.
// -------------------------------------------------------------------------

// 1-D TDM copy: global -> LDS, nbytes must be a multiple of 8.
// D# built per CDNA5 ISA ch.8: group0 = {count/type/addr}, group1 = dims.
// This toolchain exposes the 6-arg builtin: (g0, g1, g2, g3, g4, cpol).
static __device__ __forceinline__ void tdm_load_1d(unsigned ldsOff,
                                                   const void* gptr,
                                                   unsigned nbytes)
{
    unsigned long long ga = (unsigned long long)(size_t)gptr;
    unsigned n8 = nbytes >> 3;                       // 8-byte elements
    u32x4 g0;
    g0[0] = 1u;                                      // count=1 (valid), user D#
    g0[1] = ldsOff;                                  // lds_addr (bytes)
    g0[2] = (unsigned)(ga & 0xFFFFFFFFu);            // global_addr[31:0]
    g0[3] = (unsigned)((ga >> 32) & 0x1FFFFFFu)      // global_addr[56:32]
            | (2u << 30);                            // type = 2 ("image")
    i32x8 g1;
    g1[0] = (int)(3u << 16);                         // wg_mask=0, data_size=8B
    g1[1] = (int)((n8 & 0xFFFFu) << 16);             // tensor_dim0[15:0]
    g1[2] = (int)((n8 >> 16) & 0xFFFFu);             // tensor_dim0[31:16], dim1=0
    g1[3] = (int)((n8 & 0xFFFFu) << 16);             // tile_dim0 = n8 (<=65535)
    g1[4] = 0;                                       // tile_dim1/2 unused
    g1[5] = (int)n8;                                 // tensor_dim0_stride lo32
    g1[6] = 0;
    g1[7] = 0;
    i32x4 z4 = {0, 0, 0, 0};                         // groups 2/3 unused (1-D)
    i32x8 z8 = {0, 0, 0, 0, 0, 0, 0, 0};             // extra group unused
    __builtin_amdgcn_tensor_load_to_lds(g0, g1, z4, z4, z8, 0);
}

// ---------------- weight pack -> fragment-ordered f16 ---------------------
__global__ void pack_kernel(const float* __restrict__ w, h16* __restrict__ dst,
                            int Cout, int CinR, int CinP, int taps,
                            int Ktiles, int Ntiles)
{
    int total = Ktiles * Ntiles * 512;
    int idx = blockIdx.x * 256 + threadIdx.x;
    if (idx >= total) return;
    int hh   = idx & 15;
    int lane = (idx >> 4) & 31;
    int tile = idx >> 9;
    int nt = tile % Ntiles, kt = tile / Ntiles;
    int khi = (lane >> 4) * 8;
    int kin = (hh < 8) ? (khi + hh) : (16 + khi + (hh - 8));
    int k   = kt * 32 + kin;
    int tap = k / CinP;
    int ci  = k - tap * CinP;
    int n   = nt * 16 + (lane & 15);
    float v = 0.0f;
    if (ci < CinR && n < Cout && tap < taps)
        v = (taps == 27) ? w[((size_t)n * CinR + ci) * 27 + tap]
                         : w[(size_t)n * CinR + ci];
    dst[idx] = (h16)v;
}

// ---------------- implicit-GEMM conv via WMMA + TDM B-staging -------------
template <int KS, int LG2W, int NT, int MT>
__global__ void __launch_bounds__(256)
conv_wmma(const h16* __restrict__ x, const h16* __restrict__ wp,
          const float* __restrict__ bias, const h16* __restrict__ resid,
          h16* __restrict__ out16, float* __restrict__ out32,
          int Bn, int ctiles, int Cout)
{
    constexpr int W       = 1 << LG2W;
    constexpr int VOX     = 1 << (3 * LG2W);
    constexpr int Mtiles  = VOX >> 4;
    constexpr int Mblocks = Mtiles / MT;
    constexpr int TAPS    = KS * KS * KS;
    constexpr int CHUNK   = 8;                 // k-tiles staged per barrier

    __shared__ __align__(128) h16 smem[CHUNK * NT * 512];

    int wave = blockIdx.x * (blockDim.x >> 5) + (threadIdx.x >> 5);
    int lane = threadIdx.x & 31;
    if (wave >= Bn * Mblocks) return;          // grids are exact: never taken
    int mb = wave % Mblocks;                   // Mblocks is constexpr pow2
    int b  = wave / Mblocks;
    int mt0 = mb * MT;

    int lx  = lane & 15;
    int khi = (lane >> 4) * 8;
    int C = ctiles << 5;
    const h16* xb = x + (size_t)b * VOX * C;

    int w0[MT], h0[MT], d0[MT];
#pragma unroll
    for (int j = 0; j < MT; ++j) {
        int m = ((mt0 + j) << 4) + lx;         // A row (voxel) for this lane
        w0[j] = m & (W - 1);
        h0[j] = (m >> LG2W) & (W - 1);
        d0[j] = m >> (2 * LG2W);
    }

    v8f zero8 = {0.f, 0.f, 0.f, 0.f, 0.f, 0.f, 0.f, 0.f};
    h16 hz = (h16)0.0f;
    v8h zero8h = {hz, hz, hz, hz, hz, hz, hz, hz};
    v8f acc[MT][NT];
#pragma unroll
    for (int j = 0; j < MT; ++j)
#pragma unroll
        for (int nn = 0; nn < NT; ++nn) acc[j][nn] = zero8;

    const v16h* sB = (const v16h*)smem;
    unsigned ldsOff = (unsigned)(size_t)(void*)smem;   // low 32 bits = LDS offset

    int Ktot = TAPS * ctiles;
    int tap = 0, ct = 0;
    bool ok[MT];
    const v8h* ap[MT];

    for (int kt0 = 0; kt0 < Ktot; kt0 += CHUNK) {
        int nkt = Ktot - kt0;
        if (nkt > CHUNK) nkt = CHUNK;
        __syncthreads();                       // previous chunk fully consumed
        if (threadIdx.x < 32) {                // wave 0 drives the TDM
            tdm_load_1d(ldsOff, wp + (size_t)kt0 * NT * 512,
                        (unsigned)nkt * NT * 1024u);
            __builtin_amdgcn_s_wait_tensorcnt(0);
        }
        __syncthreads();                       // B chunk visible to all waves

        for (int q = 0; q < nkt; ++q) {
            if (ct == 0) {                     // new tap: im2col setup
                int dz = 0, dy = 0, dx = 0;
                if (KS == 3) {
                    dz = tap / 9;
                    int r = tap - dz * 9;
                    dy = r / 3;
                    dx = r - dy * 3;
                    dz -= 1; dy -= 1; dx -= 1;
                }
#pragma unroll
                for (int j = 0; j < MT; ++j) {
                    int zz = d0[j] + dz, yy = h0[j] + dy, xx = w0[j] + dx;
                    ok[j] = ((unsigned)zz < (unsigned)W) &&
                            ((unsigned)yy < (unsigned)W) &&
                            ((unsigned)xx < (unsigned)W);
                    int zc = min(max(zz, 0), W - 1);
                    int yc = min(max(yy, 0), W - 1);
                    int xc = min(max(xx, 0), W - 1);
                    int vcl = (((zc << LG2W) + yc) << LG2W) + xc;
                    ap[j] = (const v8h*)(xb + (size_t)vcl * C + khi);
                }
            }
            v16h afrag[MT];
#pragma unroll
            for (int j = 0; j < MT; ++j) {
                v8h a0 = ap[j][0];             // K khi..khi+7   (b128)
                v8h a1 = ap[j][2];             // K 16+khi..+7   (b128)
                a0 = ok[j] ? a0 : zero8h;      // zero 'SAME' padding
                a1 = ok[j] ? a1 : zero8h;
                afrag[j] = __builtin_shufflevector(a0, a1,
                    0, 1, 2, 3, 4, 5, 6, 7, 8, 9, 10, 11, 12, 13, 14, 15);
                ap[j] += 4;                    // +32 channels
            }
            v16h bfrag[NT];
#pragma unroll
            for (int nn = 0; nn < NT; ++nn)    // ds_load_b128 x2 each
                bfrag[nn] = sB[(q * NT + nn) * 32 + lane];
#pragma unroll
            for (int j = 0; j < MT; ++j)
#pragma unroll
                for (int nn = 0; nn < NT; ++nn)
                    acc[j][nn] = __builtin_amdgcn_wmma_f32_16x16x32_f16(
                        false, afrag[j], false, bfrag[nn], (short)0,
                        acc[j][nn], false, false);
            if (++ct == ctiles) { ct = 0; ++tap; }
        }
    }

    // D layout: VGPR r -> M = r + 8*(lane>>4), N = lane&15
#pragma unroll
    for (int j = 0; j < MT; ++j) {
        int mbase = ((mt0 + j) << 4) + (lane >> 4) * 8;
#pragma unroll
        for (int nn = 0; nn < NT; ++nn) {
            int n = nn * 16 + lx;
            if (n < Cout) {
                float bv = bias ? bias[n] : 0.0f;
#pragma unroll
                for (int r = 0; r < 8; ++r) {
                    int mm = mbase + r;
                    float val = acc[j][nn][r] + bv;
                    if (resid) val += (float)resid[((size_t)b * VOX + mm) * Cout + n];
                    if (out32) out32[((size_t)b * Cout + n) * VOX + mm] = val; // NCDHW f32
                    else       out16[((size_t)b * VOX + mm) * Cout + n] = (h16)val;
                }
            }
        }
    }
}

// ---------------- input: NCDHW f32 (4ch) -> NDHWC f16 (32ch, zero-pad) ----
__global__ void padx_kernel(const float* __restrict__ x, h16* __restrict__ y,
                            int CinR, int CinP, int VOX, int total)
{
    int idx = blockIdx.x * 256 + threadIdx.x;
    if (idx >= total) return;
    int c = idx % CinP;
    int rest = idx / CinP;
    int v = rest % VOX, b = rest / VOX;
    y[idx] = (c < CinR) ? (h16)x[((size_t)b * CinR + c) * VOX + v] : (h16)0.0f;
}

// ---------------- GroupNorm (+FiLM) (+SiLU), NDHWC f16 --------------------
__global__ void __launch_bounds__(256)
gn_kernel(const h16* __restrict__ x, h16* __restrict__ y,
          const float* __restrict__ gamma, const float* __restrict__ beta,
          const float* __restrict__ film, int C, int lg2VOX, int silu)
{
    int VOX = 1 << lg2VOX;
    int Cg = C / NGROUPS;
    int b = blockIdx.x / NGROUPS, g = blockIdx.x % NGROUPS;
    int c0 = g * Cg;
    size_t vb = (size_t)b * VOX;
    float s = 0.0f, sq = 0.0f;
    for (int v = threadIdx.x; v < VOX; v += 256) {
        size_t base = (vb + v) * C + c0;
        for (int j = 0; j < Cg; ++j) {
            float t = (float)x[base + j];
            s += t; sq += t * t;
        }
    }
    __shared__ float sh1[256], sh2[256];
    sh1[threadIdx.x] = s; sh2[threadIdx.x] = sq;
    __syncthreads();
    for (int off = 128; off > 0; off >>= 1) {
        if (threadIdx.x < off) {
            sh1[threadIdx.x] += sh1[threadIdx.x + off];
            sh2[threadIdx.x] += sh2[threadIdx.x + off];
        }
        __syncthreads();
    }
    float N = (float)(Cg * VOX);
    float mean = sh1[0] / N;
    float var  = sh2[0] / N - mean * mean;
    float rstd = rsqrtf(var + 1e-5f);
    for (int v = threadIdx.x; v < VOX; v += 256) {
        size_t base = (vb + v) * C + c0;
        for (int j = 0; j < Cg; ++j) {
            int c = c0 + j;
            float val = ((float)x[base + j] - mean) * rstd * gamma[c] + beta[c];
            if (film) val = val * (1.0f + film[b * 2 * C + c]) + film[b * 2 * C + C + c];
            if (silu) val = val / (1.0f + expf(-val));
            y[base + j] = (h16)val;
        }
    }
}

// ---------------- 2x avg-pool / nearest up (NDHWC f16) --------------------
__global__ void pool_kernel(const h16* __restrict__ x, h16* __restrict__ y,
                            int C, int total, int lg2Wo)
{
    int idx = blockIdx.x * 256 + threadIdx.x;
    if (idx >= total) return;
    int c = idx % C;
    int rest = idx / C;
    int VOXo = 1 << (3 * lg2Wo);
    int v = rest % VOXo, b = rest / VOXo;
    int mo = (1 << lg2Wo) - 1;
    int wo = v & mo, ho = (v >> lg2Wo) & mo, d0 = v >> (2 * lg2Wo);
    int lg2Wi = lg2Wo + 1;
    size_t vb = (size_t)b << (3 * lg2Wi);
    float s = 0.0f;
    for (int dz = 0; dz < 2; ++dz)
        for (int dy = 0; dy < 2; ++dy)
            for (int dx = 0; dx < 2; ++dx) {
                size_t vi = ((((size_t)(2 * d0 + dz) << lg2Wi) + (2 * ho + dy)) << lg2Wi)
                            + (2 * wo + dx);
                s += (float)x[(vb + vi) * C + c];
            }
    y[idx] = (h16)(s * 0.125f);
}

__global__ void up_kernel(const h16* __restrict__ x, h16* __restrict__ y,
                          int C, int total, int lg2Wo)
{
    int idx = blockIdx.x * 256 + threadIdx.x;
    if (idx >= total) return;
    int c = idx % C;
    int rest = idx / C;
    int VOXo = 1 << (3 * lg2Wo);
    int v = rest % VOXo, b = rest / VOXo;
    int mo = (1 << lg2Wo) - 1;
    int wo = v & mo, ho = (v >> lg2Wo) & mo, d0 = v >> (2 * lg2Wo);
    int lg2Wi = lg2Wo - 1;
    size_t vi = ((((size_t)(d0 >> 1) << lg2Wi) + (ho >> 1)) << lg2Wi) + (wo >> 1);
    y[idx] = x[(((size_t)b << (3 * lg2Wi)) + vi) * C + c];
}

// ---------------- channel concat (NDHWC f16) ------------------------------
__global__ void concat_kernel(const h16* __restrict__ s1, int C1,
                              const h16* __restrict__ s2, int C2,
                              h16* __restrict__ y, int VOX, int total)
{
    int idx = blockIdx.x * 256 + threadIdx.x;
    if (idx >= total) return;
    int Ct = C1 + C2;
    int c = idx % Ct;
    int rest = idx / Ct;
    int v = rest % VOX, b = rest / VOX;
    y[idx] = (c < C1) ? s1[((size_t)b * VOX + v) * C1 + c]
                      : s2[((size_t)b * VOX + v) * C2 + (c - C1)];
}

// ---------------- timestep embedding MLP ----------------------------------
__global__ void time_embed_kernel(const int* __restrict__ ts,
                                  const float* __restrict__ w1, const float* __restrict__ b1,
                                  const float* __restrict__ w2, const float* __restrict__ b2,
                                  float* __restrict__ emb)
{
    __shared__ float enc[32];
    __shared__ float hid[128];
    int b = blockIdx.x, tid = threadIdx.x;
    float t = (float)ts[b];
    if (tid < 32) {
        int i = tid & 15;
        float fr = expf(-logf(10000.0f) * (float)i / 16.0f);
        float a = t * fr;
        enc[tid] = (tid < 16) ? cosf(a) : sinf(a);
    }
    __syncthreads();
    float s = b1[tid];
    for (int i = 0; i < 32; ++i) s += w1[tid * 32 + i] * enc[i];
    hid[tid] = s / (1.0f + expf(-s));
    __syncthreads();
    float s2 = b2[tid];
    for (int i = 0; i < 128; ++i) s2 += w2[tid * 128 + i] * hid[i];
    emb[b * 128 + tid] = s2;
}

// ---------------- per-resblock FiLM projection ----------------------------
__global__ void film_kernel(const float* __restrict__ emb, const float* __restrict__ w,
                            const float* __restrict__ bias, float* __restrict__ eh, int C2)
{
    int b = blockIdx.x, o = threadIdx.x;
    if (o >= C2) return;
    float s = bias[o];
    for (int i = 0; i < 128; ++i) {
        float e = emb[b * 128 + i];
        s += w[o * 128 + i] * (e / (1.0f + expf(-e)));
    }
    eh[b * C2 + o] = s;
}

// =========================================================================
extern "C" void kernel_launch(void* const* d_in, const int* in_sizes, int n_in,
                              void* d_out, int out_size, void* d_ws, size_t ws_size,
                              hipStream_t stream)
{
    (void)in_sizes; (void)n_in; (void)out_size; (void)ws_size;
    const float* x  = (const float*)d_in[0];
    const int*   ts = (const int*)d_in[1];

    int pi = 2;
    auto P = [&](int i) -> const float* { return (const float*)d_in[i]; };
    int i_te_w1 = pi++, i_te_b1 = pi++, i_te_w2 = pi++, i_te_b2 = pi++;
    int i_in_w  = pi++, i_in_b  = pi++;

    struct RB { int gn1_g, gn1_b, w1, b1, emb_w, emb_b, gn2_g, gn2_b, w2, b2, ws, bs;
                int cin, cout; };
    auto takeRB = [&](int cin, int cout) -> RB {
        RB r;
        r.gn1_g = pi++; r.gn1_b = pi++; r.w1 = pi++; r.b1 = pi++;
        r.emb_w = pi++; r.emb_b = pi++; r.gn2_g = pi++; r.gn2_b = pi++;
        r.w2 = pi++; r.b2 = pi++;
        if (cin != cout) { r.ws = pi++; r.bs = pi++; } else { r.ws = -1; r.bs = -1; }
        r.cin = cin; r.cout = cout; return r;
    };
    RB enc1 = takeRB(32, 32),  enc2 = takeRB(32, 32),  enc3 = takeRB(32, 64);
    RB mid1 = takeRB(64, 64),  mid2 = takeRB(64, 64);
    RB dec0 = takeRB(128, 64), dec1a = takeRB(96, 64), dec1b = takeRB(64, 64);
    RB dec2 = takeRB(96, 32),  dec3 = takeRB(64, 32);
    int i_out_g = pi++, i_out_b = pi++, i_out_w = pi++, i_out_bias = pi++;

    // ---- workspace bump allocator (f16 activations: ~85 MB total) ----
    size_t cur = 0;
    auto alloc = [&](size_t bytes) -> void* {
        cur = (cur + 255) & ~(size_t)255;
        void* p = (char*)d_ws + cur;
        cur += bytes;
        return p;
    };
    const int Bn = 2;
    float* EMB = (float*)alloc((size_t)Bn * 128 * 4);
    float* EH  = (float*)alloc((size_t)Bn * 256 * 4);
    h16* PADX = (h16*)alloc((size_t)Bn * 32 * 32768 * 2);
    h16* HS0  = (h16*)alloc((size_t)Bn * 32 * 32768 * 2);
    h16* HS1  = (h16*)alloc((size_t)Bn * 32 * 32768 * 2);
    h16* HS2  = (h16*)alloc((size_t)Bn * 32 * 4096 * 2);
    h16* HS3  = (h16*)alloc((size_t)Bn * 64 * 4096 * 2);
    h16* CURA = (h16*)alloc((size_t)Bn * 96 * 32768 * 2);
    h16* CURB = (h16*)alloc((size_t)Bn * 96 * 32768 * 2);
    h16* T1 = (h16*)alloc((size_t)Bn * 96 * 32768 * 2);
    h16* TP = (h16*)alloc((size_t)Bn * 64 * 32768 * 2);
    h16* XS = (h16*)alloc((size_t)Bn * 64 * 32768 * 2);
    h16* HB = (h16*)alloc((size_t)Bn * 64 * 32768 * 2);
    h16* T2 = (h16*)alloc((size_t)Bn * 64 * 32768 * 2);

    // ---- weight packing (deterministic, each launch) ----
    auto packConv = [&](int widx, int cinR, int cinP, int cout, int ks) -> h16* {
        int taps = ks * ks * ks;
        int Kt = taps * (cinP >> 5);
        int Nt = (cout + 15) / 16;
        int total = Kt * Nt * 512;
        h16* p = (h16*)alloc((size_t)total * 2);
        pack_kernel<<<(total + 255) / 256, 256, 0, stream>>>(P(widx), p, cout,
                                                             cinR, cinP, taps, Kt, Nt);
        return p;
    };
    struct Packs { h16 *w1, *w2, *ws; };
    auto packRB = [&](const RB& r) -> Packs {
        Packs pk;
        pk.w1 = packConv(r.w1, r.cin, r.cin, r.cout, 3);
        pk.w2 = packConv(r.w2, r.cout, r.cout, r.cout, 3);
        pk.ws = (r.ws >= 0) ? packConv(r.ws, r.cin, r.cin, r.cout, 1) : nullptr;
        return pk;
    };
    h16* pk_in  = packConv(i_in_w, 4, 32, 32, 3);   // Cin zero-padded 4 -> 32
    Packs pEnc1 = packRB(enc1), pEnc2 = packRB(enc2), pEnc3 = packRB(enc3);
    Packs pMid1 = packRB(mid1), pMid2 = packRB(mid2);
    Packs pDec0 = packRB(dec0), pDec1a = packRB(dec1a), pDec1b = packRB(dec1b);
    Packs pDec2 = packRB(dec2), pDec3 = packRB(dec3);
    h16* pk_out = packConv(i_out_w, 32, 32, 4, 3);

    // ---- op launchers ----
    auto conv = [&](const h16* in, const h16* wp, const float* bias, const h16* resid,
                    h16* out16, float* out32, int cinP, int cout, int lg2W, int ks) {
        int MTv = (lg2W == 5) ? 2 : 1;         // M-blocking on the big layers
        int Mt = (1 << (3 * lg2W)) >> 4;
        int waves = Bn * (Mt / MTv);
        int blocks = (waves + 7) / 8;
        int Nt = (cout + 15) / 16;
        int ct = cinP >> 5;
#define LAUNCH_CONV(KS_, LW_, NT_, MT_) \
        conv_wmma<KS_, LW_, NT_, MT_><<<blocks, 256, 0, stream>>>( \
            in, wp, bias, resid, out16, out32, Bn, ct, cout)
        if (ks == 3) {
            if (lg2W == 5) {
                if (Nt == 1)      LAUNCH_CONV(3, 5, 1, 2);
                else if (Nt == 2) LAUNCH_CONV(3, 5, 2, 2);
                else              LAUNCH_CONV(3, 5, 4, 2);
            } else {
                if (Nt == 2)      LAUNCH_CONV(3, 4, 2, 1);
                else              LAUNCH_CONV(3, 4, 4, 1);
            }
        } else {
            if (lg2W == 5)        LAUNCH_CONV(1, 5, 2, 2);
            else                  LAUNCH_CONV(1, 4, 4, 1);
        }
#undef LAUNCH_CONV
    };
    auto gn = [&](const h16* xin, h16* y, int C, int lg2W, int gi, int bi,
                  const float* film, int silu) {
        gn_kernel<<<Bn * NGROUPS, 256, 0, stream>>>(xin, y, P(gi), P(bi), film,
                                                    C, 3 * lg2W, silu);
    };
    auto concat = [&](const h16* s1, int C1, const h16* s2, int C2, h16* y, int lg2W) {
        int VOX = 1 << (3 * lg2W);
        int total = Bn * (C1 + C2) * VOX;
        concat_kernel<<<(total + 255) / 256, 256, 0, stream>>>(s1, C1, s2, C2, y, VOX, total);
    };

    // mode: 0 = plain, 1 = down(avg-pool 2x), 2 = up(nearest 2x)
    auto resblock = [&](const h16* xin, h16* out, int lg2Win, int mode,
                        const RB& r, const Packs& pk) {
        int cin = r.cin, cout = r.cout;
        gn(xin, T1, cin, lg2Win, r.gn1_g, r.gn1_b, nullptr, 1);
        const h16* hin; const h16* xs; int lg2Wo;
        if (mode == 1) {
            lg2Wo = lg2Win - 1;
            int tot = Bn * cin * (1 << (3 * lg2Wo));
            pool_kernel<<<(tot + 255) / 256, 256, 0, stream>>>(T1, TP, cin, tot, lg2Wo);
            pool_kernel<<<(tot + 255) / 256, 256, 0, stream>>>(xin, XS, cin, tot, lg2Wo);
            hin = TP; xs = XS;
        } else if (mode == 2) {
            lg2Wo = lg2Win + 1;
            int tot = Bn * cin * (1 << (3 * lg2Wo));
            up_kernel<<<(tot + 255) / 256, 256, 0, stream>>>(T1, TP, cin, tot, lg2Wo);
            up_kernel<<<(tot + 255) / 256, 256, 0, stream>>>(xin, XS, cin, tot, lg2Wo);
            hin = TP; xs = XS;
        } else {
            hin = T1; xs = xin; lg2Wo = lg2Win;
        }
        conv(hin, pk.w1, P(r.b1), nullptr, HB, nullptr, cin, cout, lg2Wo, 3);
        film_kernel<<<Bn, 128, 0, stream>>>(EMB, P(r.emb_w), P(r.emb_b), EH, 2 * cout);
        gn(HB, T2, cout, lg2Wo, r.gn2_g, r.gn2_b, EH, 1);
        const h16* resid = xs;
        if (r.ws >= 0) {                       // 1x1 skip conv; HB is dead now
            conv(xs, pk.ws, P(r.bs), nullptr, HB, nullptr, cin, cout, lg2Wo, 1);
            resid = HB;
        }
        conv(T2, pk.w2, P(r.b2), resid, out, nullptr, cout, cout, lg2Wo, 3);
    };

    // ---- network ----
    time_embed_kernel<<<Bn, 128, 0, stream>>>(ts, P(i_te_w1), P(i_te_b1),
                                              P(i_te_w2), P(i_te_b2), EMB);
    {   // NCDHW f32 input -> NDHWC f16, channels zero-padded 4 -> 32
        int tot = Bn * 32 * 32768;
        padx_kernel<<<(tot + 255) / 256, 256, 0, stream>>>(x, PADX, 4, 32, 32768, tot);
    }
    conv(PADX, pk_in, P(i_in_b), nullptr, HS0, nullptr, 32, 32, 5, 3); // h0: 32 @ 32^3
    resblock(HS0, HS1, 5, 0, enc1, pEnc1);                         // h1: 32 @ 32^3
    resblock(HS1, HS2, 5, 1, enc2, pEnc2);                         // h2: 32 @ 16^3
    resblock(HS2, HS3, 4, 0, enc3, pEnc3);                         // h3: 64 @ 16^3
    resblock(HS3, CURA, 4, 0, mid1, pMid1);
    resblock(CURA, CURB, 4, 0, mid2, pMid2);
    concat(HS3, 64, CURB, 64, CURA, 4);                            // 128 @ 16^3
    resblock(CURA, CURB, 4, 0, dec0, pDec0);                       // 64 @ 16^3
    concat(HS2, 32, CURB, 64, CURA, 4);                            // 96 @ 16^3
    resblock(CURA, CURB, 4, 0, dec1a, pDec1a);                     // 64 @ 16^3
    resblock(CURB, CURA, 4, 2, dec1b, pDec1b);                     // 64 @ 32^3
    concat(HS1, 32, CURA, 64, CURB, 5);                            // 96 @ 32^3
    resblock(CURB, CURA, 5, 0, dec2, pDec2);                       // 32 @ 32^3
    concat(HS0, 32, CURA, 32, CURB, 5);                            // 64 @ 32^3
    resblock(CURB, CURA, 5, 0, dec3, pDec3);                       // 32 @ 32^3
    gn(CURA, T1, 32, 5, i_out_g, i_out_b, nullptr, 1);
    conv(T1, pk_out, P(i_out_bias), nullptr, nullptr, (float*)d_out, 32, 4, 5, 3);
}
